// JointAttention_89824946028755
// MI455X (gfx1250) — matrix operand order repaired and checked by reading, not compiled, not run on hardware
//
#include <hip/hip_runtime.h>
#include <hip/hip_bf16.h>
#include <math.h>

// Joint attention for MI455X (gfx1250, wave32, WMMA).
// Pipeline: 6x bf16-WMMA GEMM (qkv, writing bf16 head-major / d-major bufs)
// -> rope fixup (bf16) -> flash attention (double-buffered async LDS staging
// via global_load_async_to_lds_b128, bf16 WMMA QK^T and PV, f32 online
// softmax) -> 2x bf16-WMMA GEMM (out proj, f32 out).

#define HEADS 16
#define HD 64

typedef __attribute__((ext_vector_type(16))) __bf16 v16bf;
typedef __attribute__((ext_vector_type(8)))  float  v8f;
typedef __attribute__((ext_vector_type(4)))  unsigned int v4u;

union Frag {
  v16bf bf;
  v4u   u4[2];
  unsigned int u[8];
};

__device__ __forceinline__ v8f wmma_bf16(v16bf a, v16bf b, v8f c) {
  return __builtin_amdgcn_wmma_f32_16x16x32_bf16(false, a, false, b, (short)0, c, false, false);
}

__device__ __forceinline__ unsigned pack_bf16(float lo, float hi) {
  union { __bf16 b[2]; unsigned u; } t;
  t.b[0] = (__bf16)lo;
  t.b[1] = (__bf16)hi;
  return t.u;
}

__device__ __forceinline__ __bf16 f2bf(float f) { return (__bf16)f; }

// --- async VMEM->LDS (CDNA5) via inline asm (portable across toolchains) ---
// Per lane: copies 16B from its global address to its LDS address; tracked by
// ASYNCcnt. LDS byte address = low 32 bits of the generic shared pointer.
__device__ __forceinline__ void ja_async_b128(const void* g, void* l) {
  unsigned lds = (unsigned)(uintptr_t)l;
  unsigned long long ga = (unsigned long long)(uintptr_t)g;
  asm volatile("global_load_async_to_lds_b128 %0, %1, off"
               :: "v"(lds), "v"(ga) : "memory");
}
__device__ __forceinline__ void ja_async_wait() {
  asm volatile("s_wait_asynccnt 0x0" ::: "memory");
}

// ---------------------------------------------------------------------------
// Generic GEMM: out = A @ W + bias (A, W f32; WMMA in bf16).
// A rows: A[b*a_bstride + (s+a_s_off)*K + k]; 256-row tiles never straddle a
// batch boundary (256 | rows_per_batch for all call sites).
// OUT_MODE 0: f32 out[row*N + col], optional mask-zeroing per (b,s).
// OUT_MODE 1: bf16 scatter, head-major [B, HEADS, s_total, HD] at s+s_off.
// OUT_MODE 2: bf16 scatter, d-major   [B, HEADS, HD, s_total] at s+s_off (V).
// Tile: 256(M) x 64(N), K-step 32; 256 threads = 8 waves, 32 rows per wave.
// ---------------------------------------------------------------------------
template <int OUT_MODE>
__global__ __launch_bounds__(256)
void ja_gemm_wmma(const float* __restrict__ A, int rows_per_batch, long a_bstride, int a_s_off,
                  const float* __restrict__ W, const float* __restrict__ bias,
                  int M, int N, int K,
                  void* __restrict__ out_p,
                  int s_total, int s_off,
                  const unsigned char* __restrict__ mask)
{
  __shared__ __align__(16) unsigned Alds[256][16];  // [row][kpair]
  __shared__ __align__(16) unsigned Wlds[64][16];   // [col][kpair] (W tile transposed)

  const int tid   = threadIdx.x;
  const int wave  = tid >> 5;
  const int lane  = tid & 31;
  const int lhalf = lane >> 4;
  const int lmod  = lane & 15;

  const int m0 = blockIdx.x * 256;
  const int n0 = blockIdx.y * 64;

  const int b_blk = m0 / rows_per_batch;            // block-uniform
  const int s_blk = m0 - b_blk * rows_per_batch;
  const float* Abase = A + (long)b_blk * a_bstride + (long)(s_blk + a_s_off) * K;

  v8f acc[2][4] = {};

  for (int k0 = 0; k0 < K; k0 += 32) {
    // Stage A tile: 256 rows x 16 k-pairs
    #pragma unroll
    for (int i = 0; i < 16; ++i) {
      int idx = tid + i * 256;          // 4096 pairs
      int r = idx >> 4, p = idx & 15;
      float2 f = *(const float2*)(Abase + (long)r * K + (k0 + 2 * p));
      Alds[r][p] = pack_bf16(f.x, f.y);
    }
    // Stage W tile transposed: 64 cols x 16 k-pairs (n fastest for coalescing)
    #pragma unroll
    for (int i = 0; i < 4; ++i) {
      int idx = tid + i * 256;          // 1024 pairs
      int p = idx >> 6, n = idx & 63;
      float w0 = W[(long)(k0 + 2 * p)     * N + (n0 + n)];
      float w1 = W[(long)(k0 + 2 * p + 1) * N + (n0 + n)];
      Wlds[n][p] = pack_bf16(w0, w1);
    }
    __syncthreads();

    Frag af[2];
    #pragma unroll
    for (int mt = 0; mt < 2; ++mt) {
      const v4u* ar = (const v4u*)&Alds[wave * 32 + mt * 16 + lmod][0];
      af[mt].u4[0] = ar[lhalf];
      af[mt].u4[1] = ar[2 + lhalf];
    }
    #pragma unroll
    for (int nt = 0; nt < 4; ++nt) {
      Frag bfr;
      const v4u* br = (const v4u*)&Wlds[nt * 16 + lmod][0];
      bfr.u4[0] = br[lhalf];
      bfr.u4[1] = br[2 + lhalf];
      acc[0][nt] = wmma_bf16(af[0].bf, bfr.bf, acc[0][nt]);
      acc[1][nt] = wmma_bf16(af[1].bf, bfr.bf, acc[1][nt]);
    }
    __syncthreads();
  }

  // Epilogue (single path per instantiation)
  #pragma unroll
  for (int mt = 0; mt < 2; ++mt) {
    #pragma unroll
    for (int nt = 0; nt < 4; ++nt) {
      int col = n0 + nt * 16 + lmod;
      float bv = bias[col];
      #pragma unroll
      for (int r = 0; r < 8; ++r) {
        int lrow = wave * 32 + mt * 16 + r + (lhalf << 3);
        int s    = s_blk + lrow;
        float val = acc[mt][nt][r] + bv;
        if (OUT_MODE == 0) {
          if (mask && !mask[b_blk * rows_per_batch + s]) val = 0.0f;
          ((float*)out_p)[(long)(m0 + lrow) * N + col] = val;
        } else if (OUT_MODE == 1) {
          int h = col >> 6, d = col & 63;
          ((__bf16*)out_p)[(((long)(b_blk * HEADS + h) * s_total) + (s + s_off)) * HD + d] = f2bf(val);
        } else {
          int h = col >> 6, d = col & 63;
          ((__bf16*)out_p)[(((long)(b_blk * HEADS + h) * HD) + d) * s_total + (s + s_off)] = f2bf(val);
        }
      }
    }
  }
}

// ---------------------------------------------------------------------------
// RoPE applies only to channels 0..63 of the 1024-wide projection == head 0.
// out[2j]   = t[2j]*cos[2j]     - t[2j+1]*sin[2j]
// out[2j+1] = t[2j+1]*cos[2j+1] + t[2j]*sin[2j+1]
// Buffers are bf16 head-major.
// ---------------------------------------------------------------------------
__global__ void ja_rope(__bf16* __restrict__ qb, __bf16* __restrict__ kb,
                        const float* __restrict__ freqs, const float* __restrict__ c_freqs,
                        int S, int NX)
{
  int bs = blockIdx.x;
  int b = bs / S, s = bs - b * S;
  int j = threadIdx.x;           // 0..31
  int d0 = 2 * j, d1 = 2 * j + 1;
  const float* fr = (s < NX) ? (freqs + (long)s * 64) : (c_freqs + (long)(s - NX) * 64);
  float c0 = __cosf(fr[d0]), s0 = __sinf(fr[d0]);
  float c1 = __cosf(fr[d1]), s1 = __sinf(fr[d1]);
  long base = ((long)(b * HEADS + 0) * S + s) * HD;   // head 0 only
  float qe = (float)qb[base + d0], qo = (float)qb[base + d1];
  qb[base + d0] = f2bf(qe * c0 - qo * s0);
  qb[base + d1] = f2bf(qo * c1 + qe * s1);
  float ke = (float)kb[base + d0], ko = (float)kb[base + d1];
  kb[base + d0] = f2bf(ke * c0 - ko * s0);
  kb[base + d1] = f2bf(ko * c1 + ke * s1);
}

// ---------------------------------------------------------------------------
// Flash attention: block = (b,h) x 128 queries; 8 waves x 16 query rows each.
// q/k are bf16 [B,H,S,HD]; v is bf16 [B,H,HD,S] (d-major), so K- and V-tile
// staging are pure 16B copies -> GLOBAL_LOAD_ASYNC_TO_LDS_B128, double-
// buffered so the DMA of tile i+1 overlaps compute of tile i. Every fragment
// is two ds_load_b128. f32 online softmax with shfl_xor row reductions inside
// 16-lane halves (C-fragment row layout m = r + 8*(lane>>4)).
// ---------------------------------------------------------------------------
__global__ __launch_bounds__(256)
void ja_flash_attn(const __bf16* __restrict__ qb, const __bf16* __restrict__ kb,
                   const __bf16* __restrict__ vb, float* __restrict__ ob,
                   const unsigned char* __restrict__ mask,
                   int S, int NX)
{
  __shared__ __align__(16) unsigned Klds[2][32][32];  // [buf][key][dpair]
  __shared__ __align__(16) unsigned Vlds[2][64][16];  // [buf][d][keypair]
  __shared__ __align__(16) __bf16   Plds[8][16][32];  // per-wave P scratch

  const int tid   = threadIdx.x;
  const int wave  = tid >> 5;
  const int lane  = tid & 31;
  const int lhalf = lane >> 4;
  const int lmod  = lane & 15;

  const int bh = blockIdx.y;
  const int b  = bh / HEADS;
  const int h  = bh - b * HEADS;
  const int q0 = blockIdx.x * 128;

  const long head_base = (long)bh * S * HD;   // same element count for q/k and v
  const __bf16* kb_head = kb + head_base;
  const __bf16* vb_head = vb + head_base;

  // per-thread staging geometry (16B per thread per tile)
  const int vd = tid >> 2, vc = tid & 3;

  // Q fragments straight from global bf16 (row = 64 bf16 = 8 v4u)
  Frag qa0, qa1;
  {
    const v4u* qr = (const v4u*)(qb + head_base + (long)(q0 + wave * 16 + lmod) * HD);
    qa0.u4[0] = qr[lhalf];
    qa0.u4[1] = qr[2 + lhalf];
    qa1.u4[0] = qr[4 + lhalf];
    qa1.u4[1] = qr[6 + lhalf];
  }

  float mrow[8], lrow[8];
  #pragma unroll
  for (int r = 0; r < 8; ++r) { mrow[r] = -1e30f; lrow[r] = 0.0f; }
  v8f acc[4] = {};
  const float scale = 0.125f;   // 1/sqrt(64)

  // prologue: issue async copy of tile 0 into buffer 0
  ja_async_b128(kb_head + (long)tid * 8,            &Klds[0][0][0] + tid * 4);
  ja_async_b128(vb_head + (long)vd * S + vc * 8,    &Vlds[0][vd][vc * 4]);

  const int nblk = S / 32;
  for (int i = 0; i < nblk; ++i) {
    const int cur  = i & 1;
    const int kblk = i * 32;

    ja_async_wait();       // tile i resident (this thread's copies)
    __syncthreads();       // all threads' copies visible; iter i-1 reads done

    if (i + 1 < nblk) {    // issue tile i+1 into the other buffer (overlaps compute)
      int nb = kblk + 32;
      ja_async_b128(kb_head + (long)nb * HD + (long)tid * 8, &Klds[cur ^ 1][0][0] + tid * 4);
      ja_async_b128(vb_head + (long)vd * S + nb + vc * 8,    &Vlds[cur ^ 1][vd][vc * 4]);
    }

    // Scores: two 16-key tiles, contraction over d=64 (two WMMAs each)
    v8f sc[2];
    float kbias[2];
    #pragma unroll
    for (int t = 0; t < 2; ++t) {
      Frag kf0, kf1;
      const v4u* kr = (const v4u*)&Klds[cur][t * 16 + lmod][0];   // 8 v4u per row
      kf0.u4[0] = kr[lhalf];
      kf0.u4[1] = kr[2 + lhalf];
      kf1.u4[0] = kr[4 + lhalf];
      kf1.u4[1] = kr[6 + lhalf];
      v8f z = {};
      z = wmma_bf16(qa0.bf, kf0.bf, z);
      sc[t] = wmma_bf16(qa1.bf, kf1.bf, z);
      int kg = kblk + t * 16 + lmod;
      bool ok = (kg >= NX) || (mask[b * NX + kg] != 0);
      kbias[t] = ok ? 0.0f : -3.0e38f;
    }

    // Online softmax per row (row of reg r = r + 8*lhalf, cols across 16 lanes)
    #pragma unroll
    for (int r = 0; r < 8; ++r) {
      float s0v = sc[0][r] * scale + kbias[0];
      float s1v = sc[1][r] * scale + kbias[1];
      float rm = fmaxf(s0v, s1v);
      rm = fmaxf(rm, __shfl_xor(rm, 8, 32));
      rm = fmaxf(rm, __shfl_xor(rm, 4, 32));
      rm = fmaxf(rm, __shfl_xor(rm, 2, 32));
      rm = fmaxf(rm, __shfl_xor(rm, 1, 32));
      float mnew = fmaxf(mrow[r], rm);
      float al = __expf(mrow[r] - mnew);
      float p0 = __expf(s0v - mnew);
      float p1 = __expf(s1v - mnew);
      float rs = p0 + p1;
      rs += __shfl_xor(rs, 8, 32);
      rs += __shfl_xor(rs, 4, 32);
      rs += __shfl_xor(rs, 2, 32);
      rs += __shfl_xor(rs, 1, 32);
      lrow[r] = lrow[r] * al + rs;
      mrow[r] = mnew;
      #pragma unroll
      for (int dt = 0; dt < 4; ++dt) acc[dt][r] *= al;
      int prow = r + (lhalf << 3);
      Plds[wave][prow][lmod]      = f2bf(p0);
      Plds[wave][prow][16 + lmod] = f2bf(p1);
    }

    // P as A-fragment (16 rows x 32 keys); same-wave LDS ops stay in order
    Frag pa;
    const v4u* pr = (const v4u*)&Plds[wave][lmod][0];        // 4 v4u per row
    pa.u4[0] = pr[lhalf];
    pa.u4[1] = pr[2 + lhalf];

    // P x V over 4 d-tiles
    #pragma unroll
    for (int dt = 0; dt < 4; ++dt) {
      Frag vf;
      const v4u* vr = (const v4u*)&Vlds[cur][dt * 16 + lmod][0];  // 4 v4u per row
      vf.u4[0] = vr[lhalf];
      vf.u4[1] = vr[2 + lhalf];
      acc[dt] = wmma_bf16(pa.bf, vf.bf, acc[dt]);
    }
  }

  // Normalize, write o in (B, S, HEADS*HD) f32 layout for the output GEMMs
  #pragma unroll
  for (int dt = 0; dt < 4; ++dt) {
    #pragma unroll
    for (int r = 0; r < 8; ++r) {
      int srow = q0 + wave * 16 + r + (lhalf << 3);
      int col  = h * HD + dt * 16 + lmod;
      ob[((long)b * S + srow) * (HEADS * HD) + col] = acc[dt][r] / lrow[r];
    }
  }
}

// ---------------------------------------------------------------------------
extern "C" void kernel_launch(void* const* d_in, const int* in_sizes, int n_in,
                              void* d_out, int out_size, void* d_ws, size_t ws_size,
                              hipStream_t stream) {
  const float* x       = (const float*)d_in[0];
  const float* c       = (const float*)d_in[1];
  const unsigned char* mask = (const unsigned char*)d_in[2];
  const float* freqs   = (const float*)d_in[3];
  const float* c_freqs = (const float*)d_in[4];
  const float* wq  = (const float*)d_in[5];  const float* bq  = (const float*)d_in[6];
  const float* wk  = (const float*)d_in[7];  const float* bk  = (const float*)d_in[8];
  const float* wv  = (const float*)d_in[9];  const float* bv  = (const float*)d_in[10];
  const float* wqc = (const float*)d_in[11]; const float* bqc = (const float*)d_in[12];
  const float* wkc = (const float*)d_in[13]; const float* bkc = (const float*)d_in[14];
  const float* wvc = (const float*)d_in[15]; const float* bvc = (const float*)d_in[16];
  const float* wo  = (const float*)d_in[17]; const float* bo  = (const float*)d_in[18];
  const float* woc = (const float*)d_in[19]; const float* boc = (const float*)d_in[20];

  const int Bn = 2, NX = 2048, NC = 256, DIM = 1024, INNER = 1024;
  const int S = NX + NC;                      // 2304
  const size_t head_elems = (size_t)Bn * HEADS * S * HD;  // 4,718,592

  // workspace: 3 bf16 qkv buffers + 1 f32 o buffer
  __bf16* q_buf = (__bf16*)d_ws;
  __bf16* k_buf = q_buf + head_elems;
  __bf16* v_buf = k_buf + head_elems;                 // d-major [B,H,HD,S]
  float*  o_buf = (float*)(v_buf + head_elems);       // [B, S, INNER]

  dim3 blk(256);

  // x projections: M = Bn*NX = 4096
  {
    dim3 g((Bn * NX) / 256, INNER / 64);
    ja_gemm_wmma<1><<<g, blk, 0, stream>>>(x, NX, (long)NX * DIM, 0, wq, bq,
                                           Bn * NX, INNER, DIM, q_buf, S, 0, nullptr);
    ja_gemm_wmma<1><<<g, blk, 0, stream>>>(x, NX, (long)NX * DIM, 0, wk, bk,
                                           Bn * NX, INNER, DIM, k_buf, S, 0, nullptr);
    ja_gemm_wmma<2><<<g, blk, 0, stream>>>(x, NX, (long)NX * DIM, 0, wv, bv,
                                           Bn * NX, INNER, DIM, v_buf, S, 0, nullptr);
  }
  // c projections: M = Bn*NC = 512, scattered at sequence offset NX
  {
    dim3 g((Bn * NC) / 256, INNER / 64);
    ja_gemm_wmma<1><<<g, blk, 0, stream>>>(c, NC, (long)NC * DIM, 0, wqc, bqc,
                                           Bn * NC, INNER, DIM, q_buf, S, NX, nullptr);
    ja_gemm_wmma<1><<<g, blk, 0, stream>>>(c, NC, (long)NC * DIM, 0, wkc, bkc,
                                           Bn * NC, INNER, DIM, k_buf, S, NX, nullptr);
    ja_gemm_wmma<2><<<g, blk, 0, stream>>>(c, NC, (long)NC * DIM, 0, wvc, bvc,
                                           Bn * NC, INNER, DIM, v_buf, S, NX, nullptr);
  }

  // RoPE on head 0 of q and k (bf16 in-place)
  ja_rope<<<Bn * S, 32, 0, stream>>>(q_buf, k_buf, freqs, c_freqs, S, NX);

  // Attention
  ja_flash_attn<<<dim3(S / 128, Bn * HEADS), blk, 0, stream>>>(q_buf, k_buf, v_buf, o_buf,
                                                               mask, S, NX);

  // Output projections into d_out (x_out then c_out, flat)
  float* x_out = (float*)d_out;
  float* c_out = x_out + (size_t)Bn * NX * DIM;
  ja_gemm_wmma<0><<<dim3((Bn * NX) / 256, DIM / 64), blk, 0, stream>>>(
      o_buf, NX, (long)S * INNER, 0, wo, bo, Bn * NX, DIM, INNER, x_out, 0, 0, mask);
  ja_gemm_wmma<0><<<dim3((Bn * NC) / 256, DIM / 64), blk, 0, stream>>>(
      o_buf, NC, (long)S * INNER, NX, woc, boc, Bn * NC, DIM, INNER, c_out, 0, 0, nullptr);
}
